// StateNetwork_63496796504813
// MI455X (gfx1250) — matrix operand order repaired and checked by reading, not compiled, not run on hardware
//
#include <hip/hip_runtime.h>

// ---------------------------------------------------------------------------
// MI455X (gfx1250): GAT conv + 2 transformer encoder layers.
// Dense math: v_wmma_f32_16x16x32_f16 (f32 accumulate).
// GEMM tile staging: Tensor Data Mover (tensor_load_to_lds) -> LDS f32 tiles,
// f32->f16 conversion at fragment-load time. s_wait_tensorcnt for completion.
// ---------------------------------------------------------------------------

typedef __attribute__((ext_vector_type(16))) _Float16     v16h;
typedef __attribute__((ext_vector_type(8)))  float        v8f;
typedef __attribute__((ext_vector_type(4)))  unsigned int u32x4;
typedef __attribute__((ext_vector_type(8)))  int          i32x8;
typedef __attribute__((ext_vector_type(4)))  int          i32x4;

#define N_   16384
#define E_   262144
#define F_   768
#define H_   4
#define C_   64
#define D_   256
#define NH_  8
#define HD_  32
#define S_   512
#define B_   32
#define DFF_ 1024
#define L_   2

#if __has_builtin(__builtin_amdgcn_tensor_load_to_lds) && __has_builtin(__builtin_amdgcn_s_wait_tensorcnt)
#define USE_TDM 1
#endif

// ---------------- monotonic float<->uint encoding for atomic max -----------
__device__ inline unsigned encf(float f) {
    unsigned u = __float_as_uint(f);
    return (u & 0x80000000u) ? ~u : (u | 0x80000000u);
}
__device__ inline float decf(unsigned u) {
    unsigned v = (u & 0x80000000u) ? (u & 0x7FFFFFFFu) : ~u;
    return __uint_as_float(v);
}
__device__ inline float lrelu(float x) { return x >= 0.f ? x : 0.2f * x; }

#ifdef USE_TDM
// ---------------------------------------------------------------------------
// Issue one TDM 2D tile load: tile_h rows of tile_w f32 elements, global row
// stride = row_stride elements, packed contiguously into LDS at lds_byte.
// D# packing per CDNA5 ISA (groups 0/1; 2D => remaining
// groups zero). Tensor dims set huge: tiles never clip in this problem.
// 6-arg builtin form (clang-23 / therock-10.0 toolchain).
// ---------------------------------------------------------------------------
__device__ inline void tdm_load_tile_2d(unsigned lds_byte, const float* gptr,
                                        unsigned tile_w, unsigned tile_h,
                                        unsigned row_stride)
{
    unsigned long long ga = (unsigned long long)(__SIZE_TYPE__)gptr;
    const unsigned TDIM = 0x40000000u;  // 2^30 "infinite" tensor extent

    u32x4 g0;
    g0[0] = 1u;                                    // count=1, user mode
    g0[1] = lds_byte;                              // lds_addr
    g0[2] = (unsigned)(ga & 0xFFFFFFFFu);          // global_addr[31:0]
    g0[3] = (unsigned)((ga >> 32) & 0x01FFFFFFu)   // global_addr[56:32]
          | (2u << 30);                            // type = 2 ("image")

    i32x8 g1;
    g1[0] = (int)(2u << 16);                       // mask=0, data_size=2 (4B)
    g1[1] = (int)((TDIM & 0xFFFFu) << 16);         // barrier_addr=0 | tensor_dim0.lo
    g1[2] = (int)((TDIM >> 16) | ((TDIM & 0xFFFFu) << 16)); // dim0.hi | dim1.lo
    g1[3] = (int)((TDIM >> 16) | (tile_w << 16));  // dim1.hi | tile_dim0
    g1[4] = (int)(tile_h & 0xFFFFu);               // tile_dim1 | tile_dim2=0
    g1[5] = (int)row_stride;                       // tensor_dim0_stride[31:0]
    g1[6] = 0;                                     // stride0.hi | dim1_stride.lo
    g1[7] = 0;

    i32x4 z4 = {0, 0, 0, 0};
    i32x8 z8 = {0, 0, 0, 0, 0, 0, 0, 0};
    __builtin_amdgcn_tensor_load_to_lds(g0, g1, z4, z4, z8, 0);
}
#endif

// ---------------------------------------------------------------------------
// Generic WMMA GEMM: C[M,N] = act(A[M,K] @ op(B) + bias (+residual))
//   transB=0: B is [K,N] row-major (NN).  transB=1: B is [N,K] row-major (NT).
// Block: 256 threads = 8 waves, block tile 128x64, wave tile 32x32, K-step 32.
// LDS tiles are f32 (TDM cannot convert); f16 conversion happens at fragment
// load. A: [128][32] K-contiguous. B: transB ? [64][32] N-major : [32][64].
// Assumes M%128==0, N%64==0, K%32==0 (true for all shapes here).
// ---------------------------------------------------------------------------
__global__ __launch_bounds__(256)
void gemm_wmma_f16(const float* __restrict__ A, int lda,
                   const float* __restrict__ Bm, int ldb, int transB,
                   const float* __restrict__ bias,
                   const float* __restrict__ residual,
                   float* __restrict__ C,
                   int M, int Ncols, int K, int relu)
{
    __shared__ float As[128 * 32];   // 16 KB
    __shared__ float Bs[2048];       // 8 KB  ([64][32] NT or [32][64] NN)

    const int tid   = threadIdx.x;
    const int lane  = tid & 31;
    const int wave  = tid >> 5;
    const int wm    = wave & 3;          // 4 waves along M
    const int wn    = wave >> 2;         // 2 waves along N
    const int row0  = blockIdx.y * 128;
    const int col0  = blockIdx.x * 64;
    const int mlo   = lane & 15;
    const int khalf = lane >> 4;

    v8f acc[2][2];
#pragma unroll
    for (int t = 0; t < 2; ++t)
#pragma unroll
        for (int u = 0; u < 2; ++u)
#pragma unroll
            for (int r = 0; r < 8; ++r) acc[t][u][r] = 0.f;

#ifdef USE_TDM
    const unsigned ldsA = (unsigned)(__SIZE_TYPE__)(&As[0]);
    const unsigned ldsB = (unsigned)(__SIZE_TYPE__)(&Bs[0]);
#endif

    for (int k0 = 0; k0 < K; k0 += 32) {
        __syncthreads();
#ifdef USE_TDM
        if (wave == 0) {
            tdm_load_tile_2d(ldsA, A + (size_t)row0 * lda + k0, 32u, 128u, (unsigned)lda);
            if (transB)
                tdm_load_tile_2d(ldsB, Bm + (size_t)col0 * ldb + k0, 32u, 64u, (unsigned)ldb);
            else
                tdm_load_tile_2d(ldsB, Bm + (size_t)k0 * ldb + col0, 64u, 32u, (unsigned)ldb);
            __builtin_amdgcn_s_wait_tensorcnt(0);
        }
#else
#pragma unroll
        for (int i = 0; i < 16; ++i) {          // 128*32 / 256
            int e = tid + i * 256;
            int r = e >> 5, c = e & 31;
            As[r * 32 + c] = A[(size_t)(row0 + r) * lda + k0 + c];
        }
        if (transB) {
#pragma unroll
            for (int i = 0; i < 8; ++i) {       // 64*32 / 256
                int e = tid + i * 256;
                int n = e >> 5, k = e & 31;
                Bs[n * 32 + k] = Bm[(size_t)(col0 + n) * ldb + k0 + k];
            }
        } else {
#pragma unroll
            for (int i = 0; i < 8; ++i) {
                int e = tid + i * 256;
                int k = e >> 6, n = e & 63;
                Bs[k * 64 + n] = Bm[(size_t)(k0 + k) * ldb + col0 + n];
            }
        }
#endif
        __syncthreads();

        v16h af[2], bf[2];
#pragma unroll
        for (int t = 0; t < 2; ++t) {
            int rbase = wm * 32 + t * 16 + mlo;
#pragma unroll
            for (int i = 0; i < 8; ++i) {
                // A 16x32 f16 layout: lanes 0-15 K{0-7,16-23}, lanes 16-31 K{8-15,24-31}
                int kb = (i < 4 ? 2 * i : 2 * i + 8) + khalf * 8;
                const float2 p = *(const float2*)&As[rbase * 32 + kb];  // 8B aligned
                af[t][2 * i]     = (_Float16)p.x;
                af[t][2 * i + 1] = (_Float16)p.y;
            }
        }
#pragma unroll
        for (int u = 0; u < 2; ++u) {
            int nbase = wn * 32 + u * 16 + mlo;
#pragma unroll
            for (int i = 0; i < 8; ++i) {
                // B 32x16 f16 layout: lanes 0-15 K{0-7,16-23}, lanes 16-31 K{8-15,24-31}, N = lane&15
                int kk = (i < 4 ? 2 * i : 2 * i + 8) + khalf * 8;
                float e0, e1;
                if (transB) {
                    const float2 p = *(const float2*)&Bs[nbase * 32 + kk];
                    e0 = p.x; e1 = p.y;
                } else {
                    e0 = Bs[kk * 64 + nbase];
                    e1 = Bs[(kk + 1) * 64 + nbase];
                }
                bf[u][2 * i]     = (_Float16)e0;
                bf[u][2 * i + 1] = (_Float16)e1;
            }
        }
#pragma unroll
        for (int t = 0; t < 2; ++t)
#pragma unroll
            for (int u = 0; u < 2; ++u)
                acc[t][u] = __builtin_amdgcn_wmma_f32_16x16x32_f16(
                    false, af[t], false, bf[u], (short)0, acc[t][u], false, false);
    }

#pragma unroll
    for (int t = 0; t < 2; ++t)
#pragma unroll
        for (int u = 0; u < 2; ++u)
#pragma unroll
            for (int r = 0; r < 8; ++r) {
                int row = row0 + wm * 32 + t * 16 + r + 8 * khalf;
                int col = col0 + wn * 32 + u * 16 + mlo;
                float v = acc[t][u][r] + bias[col];
                if (residual) v += residual[(size_t)row * Ncols + col];
                if (relu) v = fmaxf(v, 0.f);
                C[(size_t)row * Ncols + col] = v;
            }
}

// ---------------------------------------------------------------------------
// Flash-style attention: one wave per (graph b, head h, 16-row query tile).
// hd = 32 == WMMA K, so QK^T of a 16x16 score tile is a single WMMA.
// Online softmax; P transposed to A-layout through 1KB LDS tile; P@V = 2 WMMAs.
// qkv layout: [16384, 768] = [Q | K | V], head cols h*32..h*32+31.
// ---------------------------------------------------------------------------
__global__ __launch_bounds__(32)
void attn_flash_wmma(const float* __restrict__ qkv, float* __restrict__ attn)
{
    __shared__ _Float16 pt[16 * 32];
    const int lane  = threadIdx.x;
    const int mlo   = lane & 15;
    const int khalf = lane >> 4;
    const int qt = blockIdx.x;   // query tile 0..31
    const int hh = blockIdx.y;   // head 0..7
    const int bg = blockIdx.z;   // graph 0..31
    const int qr0 = bg * S_ + qt * 16;
    const float scale = 0.17677669529663687f;  // 1/sqrt(32), folded into Q

    // Q fragment (A layout)
    v16h qa;
#pragma unroll
    for (int i = 0; i < 8; ++i) {
        int kb = (i < 4 ? 2 * i : 2 * i + 8) + khalf * 8;
        const float2 p = *(const float2*)(qkv + (size_t)(qr0 + mlo) * 768 + hh * 32 + kb);
        qa[2 * i]     = (_Float16)(p.x * scale);
        qa[2 * i + 1] = (_Float16)(p.y * scale);
    }

    float mrow[8], lrow[8];
    v8f o0 = {}, o1 = {};
#pragma unroll
    for (int r = 0; r < 8; ++r) { mrow[r] = -1e30f; lrow[r] = 0.f; }

    for (int kb = 0; kb < S_; kb += 32) {
        // K^T fragments: B(k,n) = K[key n, dim k]
        v16h bk0, bk1;
#pragma unroll
        for (int i = 0; i < 8; ++i) {
            int kd = (i < 4 ? 2 * i : 2 * i + 8) + khalf * 8;  // dim within head
            const float2 p0 = *(const float2*)(qkv + (size_t)(bg * S_ + kb + mlo) * 768 + 256 + hh * 32 + kd);
            const float2 p1 = *(const float2*)(qkv + (size_t)(bg * S_ + kb + 16 + mlo) * 768 + 256 + hh * 32 + kd);
            bk0[2 * i] = (_Float16)p0.x; bk0[2 * i + 1] = (_Float16)p0.y;
            bk1[2 * i] = (_Float16)p1.x; bk1[2 * i + 1] = (_Float16)p1.y;
        }
        v8f z = {};
        v8f s0 = __builtin_amdgcn_wmma_f32_16x16x32_f16(false, qa, false, bk0, (short)0, z, false, false);
        v8f s1 = __builtin_amdgcn_wmma_f32_16x16x32_f16(false, qa, false, bk1, (short)0, z, false, false);

        // online softmax; D-frag row m = r + 8*khalf, col = key = lane&15 (+16)
#pragma unroll
        for (int r = 0; r < 8; ++r) {
            float mx = fmaxf(s0[r], s1[r]);
#pragma unroll
            for (int msk = 1; msk < 16; msk <<= 1) mx = fmaxf(mx, __shfl_xor(mx, msk, 32));
            float mnew = fmaxf(mrow[r], mx);
            float resc = __expf(mrow[r] - mnew);
            float p0 = __expf(s0[r] - mnew);
            float p1 = __expf(s1[r] - mnew);
            float ps = p0 + p1;
#pragma unroll
            for (int msk = 1; msk < 16; msk <<= 1) ps += __shfl_xor(ps, msk, 32);
            lrow[r] = lrow[r] * resc + ps;
            o0[r] *= resc; o1[r] *= resc;
            mrow[r] = mnew;
            int m = r + 8 * khalf;
            pt[m * 32 + mlo]      = (_Float16)p0;
            pt[m * 32 + mlo + 16] = (_Float16)p1;
        }
        __syncthreads();

        // P as A fragment (16 rows x 32 keys)
        v16h pa;
#pragma unroll
        for (int i = 0; i < 8; ++i) {
            int kk = (i < 4 ? 2 * i : 2 * i + 8) + khalf * 8;
            pa[2 * i]     = pt[mlo * 32 + kk];
            pa[2 * i + 1] = pt[mlo * 32 + kk + 1];
        }
        // V fragments: B(k,n) = V[key kb+k, dim n]
        v16h bv0, bv1;
#pragma unroll
        for (int i = 0; i < 8; ++i) {
            int kv = (i < 4 ? 2 * i : 2 * i + 8) + khalf * 8;
            const float* V0 = qkv + (size_t)(bg * S_ + kb + kv) * 768 + 512 + hh * 32;
            const float* V1 = qkv + (size_t)(bg * S_ + kb + kv + 1) * 768 + 512 + hh * 32;
            bv0[2 * i] = (_Float16)V0[mlo];      bv0[2 * i + 1] = (_Float16)V1[mlo];
            bv1[2 * i] = (_Float16)V0[mlo + 16]; bv1[2 * i + 1] = (_Float16)V1[mlo + 16];
        }
        o0 = __builtin_amdgcn_wmma_f32_16x16x32_f16(false, pa, false, bv0, (short)0, o0, false, false);
        o1 = __builtin_amdgcn_wmma_f32_16x16x32_f16(false, pa, false, bv1, (short)0, o1, false, false);
        __syncthreads();
    }

#pragma unroll
    for (int r = 0; r < 8; ++r) {
        int row = qr0 + r + 8 * khalf;
        float inv = 1.0f / lrow[r];
        attn[(size_t)row * D_ + hh * 32 + mlo]      = o0[r] * inv;
        attn[(size_t)row * D_ + hh * 32 + 16 + mlo] = o1[r] * inv;
    }
}

// ---------------------------------------------------------------------------
// LayerNorm over 256 features, one wave per row.
// ---------------------------------------------------------------------------
__global__ __launch_bounds__(256)
void layernorm_256(const float* __restrict__ X, const float* __restrict__ g,
                   const float* __restrict__ b, float* __restrict__ Y)
{
    int wave = threadIdx.x >> 5, lane = threadIdx.x & 31;
    int row = blockIdx.x * 8 + wave;
    const float* x = X + (size_t)row * D_;
    float v[8], sum = 0.f;
#pragma unroll
    for (int i = 0; i < 8; ++i) { v[i] = x[lane + 32 * i]; sum += v[i]; }
#pragma unroll
    for (int m = 1; m < 32; m <<= 1) sum += __shfl_xor(sum, m, 32);
    float mu = sum * (1.f / 256.f);
    float var = 0.f;
#pragma unroll
    for (int i = 0; i < 8; ++i) { float d = v[i] - mu; var += d * d; }
#pragma unroll
    for (int m = 1; m < 32; m <<= 1) var += __shfl_xor(var, m, 32);
    float inv = rsqrtf(var * (1.f / 256.f) + 1e-5f);
#pragma unroll
    for (int i = 0; i < 8; ++i) {
        int c = lane + 32 * i;
        Y[(size_t)row * D_ + c] = (v[i] - mu) * inv * g[c] + b[c];
    }
}

// ---------------------------------------------------------------------------
// GAT helpers
// ---------------------------------------------------------------------------
__global__ void zero_u32(unsigned* p, int n) {
    int i = blockIdx.x * blockDim.x + threadIdx.x;
    if (i < n) p[i] = 0u;
}

__global__ void gat_attn_coef(const float* __restrict__ hfeat,
                              const float* __restrict__ att_src,
                              const float* __restrict__ att_dst,
                              float* __restrict__ asrc, float* __restrict__ adst)
{
    int idx = blockIdx.x * blockDim.x + threadIdx.x;
    if (idx >= N_ * H_) return;
    int n = idx >> 2, h = idx & 3;
    float ss = 0.f, dd = 0.f;
    const float* hp = hfeat + (size_t)n * D_ + h * C_;
    for (int c = 0; c < C_; ++c) {
        float hv = hp[c];
        ss += hv * att_src[h * C_ + c];
        dd += hv * att_dst[h * C_ + c];
    }
    asrc[idx] = ss; adst[idx] = dd;
}

__device__ inline void edge_ids(int e, const int* ei, int& src, int& dst) {
    if (e < E_) { src = ei[e]; dst = ei[E_ + e]; }
    else        { src = e - E_; dst = e - E_; }          // self loops
}

__global__ void gat_pass_max(const int* __restrict__ ei, const float* __restrict__ asrc,
                             const float* __restrict__ adst, unsigned* __restrict__ mmax)
{
    int idx = blockIdx.x * blockDim.x + threadIdx.x;
    if (idx >= (E_ + N_) * H_) return;
    int e = idx >> 2, h = idx & 3, src, dst;
    edge_ids(e, ei, src, dst);
    float lg = lrelu(asrc[src * H_ + h] + adst[dst * H_ + h]);
    atomicMax(&mmax[dst * H_ + h], encf(lg));
}

__global__ void gat_pass_sum(const int* __restrict__ ei, const float* __restrict__ asrc,
                             const float* __restrict__ adst, const unsigned* __restrict__ mmax,
                             float* __restrict__ denom)
{
    int idx = blockIdx.x * blockDim.x + threadIdx.x;
    if (idx >= (E_ + N_) * H_) return;
    int e = idx >> 2, h = idx & 3, src, dst;
    edge_ids(e, ei, src, dst);
    float lg = lrelu(asrc[src * H_ + h] + adst[dst * H_ + h]);
    atomicAdd(&denom[dst * H_ + h], __expf(lg - decf(mmax[dst * H_ + h])));
}

__global__ void gat_init_bias(const float* __restrict__ bias, float* __restrict__ out)
{
    int idx = blockIdx.x * blockDim.x + threadIdx.x;
    if (idx >= N_ * D_) return;
    out[idx] = bias[idx & (D_ - 1)];
}

__global__ void gat_pass_msg(const int* __restrict__ ei, const float* __restrict__ asrc,
                             const float* __restrict__ adst, const unsigned* __restrict__ mmax,
                             const float* __restrict__ denom, const float* __restrict__ hfeat,
                             float* __restrict__ out)
{
    int idx = blockIdx.x * blockDim.x + threadIdx.x;
    if (idx >= (E_ + N_) * H_) return;
    int e = idx >> 2, h = idx & 3, src, dst;
    edge_ids(e, ei, src, dst);
    float lg = lrelu(asrc[src * H_ + h] + adst[dst * H_ + h]);
    float alpha = __expf(lg - decf(mmax[dst * H_ + h])) / (denom[dst * H_ + h] + 1e-16f);
    const float* hp = hfeat + (size_t)src * D_ + h * C_;
    float* op = out + (size_t)dst * D_ + h * C_;
    for (int c = 0; c < C_; ++c) atomicAdd(&op[c], alpha * hp[c]);
}

__global__ void gather_last(const float* __restrict__ x, float* __restrict__ out)
{
    int idx = blockIdx.x * blockDim.x + threadIdx.x;
    if (idx >= B_ * D_) return;
    int b = idx >> 8, d = idx & (D_ - 1);
    out[idx] = x[((size_t)b * S_ + (S_ - 1)) * D_ + d];
}

// ---------------------------------------------------------------------------
extern "C" void kernel_launch(void* const* d_in, const int* in_sizes, int n_in,
                              void* d_out, int out_size, void* d_ws, size_t ws_size,
                              hipStream_t stream)
{
    const float* x       = (const float*)d_in[0];
    const int*   ei      = (const int*)  d_in[1];
    const float* W_gat   = (const float*)d_in[2];
    const float* att_src = (const float*)d_in[3];
    const float* att_dst = (const float*)d_in[4];
    const float* b_gat   = (const float*)d_in[5];
    const float* Wqkv    = (const float*)d_in[6];
    const float* bqkv    = (const float*)d_in[7];
    const float* Wo      = (const float*)d_in[8];
    const float* bo      = (const float*)d_in[9];
    const float* ln1_s   = (const float*)d_in[10];
    const float* ln1_b   = (const float*)d_in[11];
    const float* W1      = (const float*)d_in[12];
    const float* b1      = (const float*)d_in[13];
    const float* W2      = (const float*)d_in[14];
    const float* b2      = (const float*)d_in[15];
    const float* ln2_s   = (const float*)d_in[16];
    const float* ln2_b   = (const float*)d_in[17];
    float* out = (float*)d_out;

    // workspace layout (floats)
    float* W = (float*)d_ws;
    size_t o = 0;
    float*    hfeat = W + o; o += (size_t)N_ * D_;
    float*    xcur  = W + o; o += (size_t)N_ * D_;   // GAT out / running activations
    float*    asrc  = W + o; o += (size_t)N_ * H_;
    float*    adst  = W + o; o += (size_t)N_ * H_;
    unsigned* mmax  = (unsigned*)(W + o); o += (size_t)N_ * H_;
    float*    denom = W + o; o += (size_t)N_ * H_;
    float*    qkv   = W + o; o += (size_t)N_ * 3 * D_;
    float*    attn  = W + o; o += (size_t)N_ * D_;
    float*    tmp   = W + o; o += (size_t)N_ * D_;
    float*    x1    = W + o; o += (size_t)N_ * D_;
    float*    ff1   = W + o; o += (size_t)N_ * DFF_;

    // ---- GAT: h = x @ W_gat + b_gat ------------------------------------
    gemm_wmma_f16<<<dim3(D_ / 64, N_ / 128), 256, 0, stream>>>(
        x, F_, W_gat, D_, /*transB=*/0, b_gat, nullptr, hfeat, N_, D_, F_, 0);

    gat_attn_coef<<<(N_ * H_ + 255) / 256, 256, 0, stream>>>(hfeat, att_src, att_dst, asrc, adst);

    zero_u32<<<(N_ * H_ * 2 + 255) / 256, 256, 0, stream>>>(mmax, N_ * H_ * 2); // mmax + denom
    const int ET = (E_ + N_) * H_;
    gat_pass_max<<<(ET + 255) / 256, 256, 0, stream>>>(ei, asrc, adst, mmax);
    gat_pass_sum<<<(ET + 255) / 256, 256, 0, stream>>>(ei, asrc, adst, mmax, denom);
    gat_init_bias<<<(N_ * D_ + 255) / 256, 256, 0, stream>>>(b_gat, xcur);
    gat_pass_msg<<<(ET + 255) / 256, 256, 0, stream>>>(ei, asrc, adst, mmax, denom, hfeat, xcur);

    // ---- Transformer layers --------------------------------------------
    for (int l = 0; l < L_; ++l) {
        const float* Wqkv_l = Wqkv + (size_t)l * 3 * D_ * D_;
        const float* bqkv_l = bqkv + (size_t)l * 3 * D_;
        const float* Wo_l   = Wo   + (size_t)l * D_ * D_;
        const float* bo_l   = bo   + (size_t)l * D_;
        const float* W1_l   = W1   + (size_t)l * DFF_ * D_;
        const float* b1_l   = b1   + (size_t)l * DFF_;
        const float* W2_l   = W2   + (size_t)l * D_ * DFF_;
        const float* b2_l   = b2   + (size_t)l * D_;

        // qkv = xcur @ Wqkv^T + bqkv  (NT: Wqkv is [3D, D])
        gemm_wmma_f16<<<dim3(3 * D_ / 64, N_ / 128), 256, 0, stream>>>(
            xcur, D_, Wqkv_l, D_, 1, bqkv_l, nullptr, qkv, N_, 3 * D_, D_, 0);

        // attention
        attn_flash_wmma<<<dim3(S_ / 16, NH_, B_), 32, 0, stream>>>(qkv, attn);

        // tmp = attn @ Wo^T + bo + xcur
        gemm_wmma_f16<<<dim3(D_ / 64, N_ / 128), 256, 0, stream>>>(
            attn, D_, Wo_l, D_, 1, bo_l, xcur, tmp, N_, D_, D_, 0);
        // x1 = LN1(tmp)
        layernorm_256<<<N_ / 8, 256, 0, stream>>>(tmp, ln1_s + (size_t)l * D_,
                                                  ln1_b + (size_t)l * D_, x1);
        // ff1 = relu(x1 @ W1^T + b1)
        gemm_wmma_f16<<<dim3(DFF_ / 64, N_ / 128), 256, 0, stream>>>(
            x1, D_, W1_l, D_, 1, b1_l, nullptr, ff1, N_, DFF_, D_, 1);
        // tmp = ff1 @ W2^T + b2 + x1
        gemm_wmma_f16<<<dim3(D_ / 64, N_ / 128), 256, 0, stream>>>(
            ff1, DFF_, W2_l, DFF_, 1, b2_l, x1, tmp, N_, D_, DFF_, 0);
        // xcur = LN2(tmp)
        layernorm_256<<<N_ / 8, 256, 0, stream>>>(tmp, ln2_s + (size_t)l * D_,
                                                  ln2_b + (size_t)l * D_, xcur);
    }

    // ---- output: last node per graph -----------------------------------
    gather_last<<<(B_ * D_ + 255) / 256, 256, 0, stream>>>(xcur, out);

    (void)in_sizes; (void)n_in; (void)out_size; (void)ws_size;
}
